// LongTermFeatureBank_4475355922728
// MI455X (gfx1250) — compile-verified, matrix operand order checked by reading, not run on hardware
//
#include <hip/hip_runtime.h>

// ---------------------------------------------------------------------------
// LongTermFeatureBank forward on MI455X (gfx1250, wave32, WMMA + TDM).
//   1. max-pool 7x7 over x[128,512,16,7,7] -> xp[128,16,512]  (HBM-bound)
//   2. per stack: fc_a/fc_b GEMMs (WMMA f16->f32, TDM-staged A tiles),
//      batched 16x16 score WMMA, batch-axis softmax, context matmul +
//      LayerNorm + ReLU, fc_1 GEMM with residual.
//   3. classifier: WMMA GEMM [128,16384]x[16384,200], tiny VALU FCs.
// ---------------------------------------------------------------------------

typedef _Float16 half_t;
typedef __attribute__((ext_vector_type(16))) _Float16     v16h;
typedef __attribute__((ext_vector_type(8)))  _Float16     v8h;
typedef __attribute__((ext_vector_type(8)))  float        v8f;
typedef __attribute__((ext_vector_type(4)))  unsigned int u32x4;
typedef __attribute__((ext_vector_type(8)))  int          i32x8;
typedef __attribute__((ext_vector_type(4)))  int          i32x4;

#define D1     512
#define D2     16
#define BATCH  128
#define M_ROWS (BATCH * D2)   // 2048
#define KCLS   (2 * D2 * D1)  // 16384
#define N1     200

#if defined(__has_include)
#if __has_include(<hip/amd_detail/amd_gfx1250_TDM.h>)
#define TDM_6ARG 1
#else
#define TDM_6ARG 0
#endif
#else
#define TDM_6ARG 0
#endif

// Build the documented 16-bit WMMA fragment (VGPR0-3: K=base..base+7,
// VGPR4-7: K=base+16..base+23, per lane-half) from two 8-half chunks.
__device__ inline v8f wmma_step(v8h a0, v8h a1, v8h b0, v8h b1, v8f acc) {
  v16h af, bf;
#pragma unroll
  for (int i = 0; i < 8; ++i) {
    af[i] = a0[i]; af[8 + i] = a1[i];
    bf[i] = b0[i]; bf[8 + i] = b1[i];
  }
  return __builtin_amdgcn_wmma_f32_16x16x32_f16(
      /*neg_a=*/false, af, /*neg_b=*/false, bf,
      /*c_mod=*/(short)0, acc, /*reuse_a=*/false, /*reuse_b=*/false);
}

// ---------------------------------------------------------------------------
// TDM: stage a 16-row x 512-half tile into LDS with padded rows.
// D# (2-D): data_size=2B, tile=[512,16], tensor_dim0_stride=512,
// pad_enable, pad_interval=256 dwords (one row), pad_amount=4 dwords
// -> LDS row stride 520 halves (matches As[16][520] bank-spread layout).
// ---------------------------------------------------------------------------
__device__ inline void tdm_load_tile_16x512(const half_t* gsrc, void* lds_dst) {
  unsigned long long ga = (unsigned long long)(uintptr_t)gsrc;
  unsigned int lds = (unsigned int)(uintptr_t)lds_dst;  // low 32b = LDS offset
  u32x4 g0;
  g0[0] = 1u;                                  // count=1 (valid user D#)
  g0[1] = lds;                                 // lds_addr
  g0[2] = (unsigned int)(ga & 0xFFFFFFFFu);    // global_addr[31:0]
  g0[3] = (unsigned int)((ga >> 32) & 0x01FFFFFFu) | 0x80000000u;  // [56:32]|type=2
  i32x8 g1;
  g1[0] = (int)0x07D10000u;  // data_size=2B | pad_en | pad_interval=256dw | pad=4dw
  g1[1] = (int)(512u << 16);                   // tensor_dim0[15:0]=512
  g1[2] = (int)((2048u & 0xFFFFu) << 16);      // tensor_dim0 hi=0 | tensor_dim1 lo=2048
  g1[3] = (int)(512u << 16);                   // tensor_dim1 hi=0 | tile_dim0=512
  g1[4] = 16;                                  // tile_dim1=16, tile_dim2=0
  g1[5] = 512;                                 // tensor_dim0_stride[31:0]=512
  g1[6] = 0;                                   // stride0 hi | tensor_dim1_stride lo
  g1[7] = 0;
  i32x4 z4 = {0, 0, 0, 0};
#if TDM_6ARG
  i32x8 z8 = {0, 0, 0, 0, 0, 0, 0, 0};
  __builtin_amdgcn_tensor_load_to_lds(g0, g1, z4, z4, z8, 0);
#else
  __builtin_amdgcn_tensor_load_to_lds(g0, g1, z4, z4, 0);
#endif
}

// ---------------------------------------------------------------------------
__global__ void k_f32_to_f16(const float* __restrict__ src,
                             half_t* __restrict__ dst, int n) {
  int i = blockIdx.x * blockDim.x + threadIdx.x;
  if (i < n) dst[i] = (half_t)src[i];
}

// ---------------------------------------------------------------------------
// Max-pool over HxW=49 (the 205 MB streaming read) + init new_x = xp
// ---------------------------------------------------------------------------
__global__ void k_pool(const float* __restrict__ x,
                       float* __restrict__ xp_f, half_t* __restrict__ xp_h,
                       float* __restrict__ nx_f, half_t* __restrict__ nx_h) {
  int o = blockIdx.x * blockDim.x + threadIdx.x;  // [0, 2048*512)
  if (o >= M_ROWS * D1) return;
  int b  = o >> 13;
  int r  = o & 8191;
  int d2 = r >> 9;
  int d1 = r & 511;
  const float* p = x + (size_t)((b * D1 + d1) * D2 + d2) * 49;
  float m = p[0];
#pragma unroll
  for (int j = 1; j < 49; ++j) m = fmaxf(m, p[j]);
  xp_f[o] = m;
  nx_f[o] = m;
  half_t h = (half_t)m;
  xp_h[o] = h;
  nx_h[o] = h;
}

// ---------------------------------------------------------------------------
// Y[M,512] = op( X[M,512] @ W[512,512]^T + bias [+ res] ).
// Block = 8 waves, one 16-row A tile staged to LDS by the TDM; each wave owns
// a 16x32 output tile (two WMMA accumulators sharing one A fragment).
// ---------------------------------------------------------------------------
__global__ void __launch_bounds__(256)
k_gemm512(const half_t* __restrict__ X, const half_t* __restrict__ W,
          const float* __restrict__ bias, const float* __restrict__ res,
          float* __restrict__ Yf, half_t* __restrict__ Yh, int relu) {
  __shared__ __align__(16) half_t As[16][520];  // 520-half rows (TDM-padded)
  const int lane = threadIdx.x & 31;
  const int wave = threadIdx.x >> 5;
  const int m0 = blockIdx.x * 16;

  if (wave == 0) tdm_load_tile_16x512(X + (size_t)m0 * D1, &As[0][0]);
  __builtin_amdgcn_s_wait_tensorcnt(0);
  __syncthreads();

  const int ntile0 = (blockIdx.y * 8 + wave) * 2;  // two adjacent N tiles
  const int lm = lane & 15, lh = lane >> 4;
  const half_t* Wrow0 = W + (size_t)(ntile0 * 16 + lm) * D1;
  const half_t* Wrow1 = Wrow0 + (size_t)16 * D1;

  v8f acc0 = {}, acc1 = {};
  for (int kk = 0; kk < D1; kk += 32) {
    if (kk + 128 < D1) {
      __builtin_prefetch(Wrow0 + kk + 128, 0, 3);
      __builtin_prefetch(Wrow1 + kk + 128, 0, 3);
    }
    int base = kk + 8 * lh;
    v8h a0 = *(const v8h*)&As[lm][base];
    v8h a1 = *(const v8h*)&As[lm][base + 16];
    v8h b0 = *(const v8h*)(Wrow0 + base);
    v8h b1 = *(const v8h*)(Wrow0 + base + 16);
    v8h c0 = *(const v8h*)(Wrow1 + base);
    v8h c1 = *(const v8h*)(Wrow1 + base + 16);
    acc0 = wmma_step(a0, a1, b0, b1, acc0);
    acc1 = wmma_step(a0, a1, c0, c1, acc1);
  }

#pragma unroll
  for (int t = 0; t < 2; ++t) {
    const v8f acc = t ? acc1 : acc0;
    const int n = (ntile0 + t) * 16 + lm;
    const float bn = bias ? bias[n] : 0.0f;
#pragma unroll
    for (int r2 = 0; r2 < 8; ++r2) {
      int m = m0 + lh * 8 + r2;  // C layout: lane->N, VGPR->M (8-row halves)
      float v = acc[r2] + bn;
      if (res) v += res[(size_t)m * D1 + n];
      if (relu) v = fmaxf(v, 0.0f);
      if (Yf) Yf[(size_t)m * D1 + n] = v;
      if (Yh) Yh[(size_t)m * D1 + n] = (half_t)v;
    }
  }
}

// ---------------------------------------------------------------------------
// scores[b,c,d] = dot(a[b,c,:], bp[b,d,:]) / sqrt(512)  — one WMMA wave per b
// ---------------------------------------------------------------------------
__global__ void __launch_bounds__(128)
k_scores(const half_t* __restrict__ A, const half_t* __restrict__ Bp,
         float* __restrict__ S) {
  const int lane = threadIdx.x & 31;
  const int wave = threadIdx.x >> 5;
  const int b = blockIdx.x * 4 + wave;
  const int lm = lane & 15, lh = lane >> 4;
  const half_t* Arow = A  + (size_t)(b * 16 + lm) * D1;
  const half_t* Brow = Bp + (size_t)(b * 16 + lm) * D1;
  v8f acc = {};
  for (int kk = 0; kk < D1; kk += 32) {
    int base = kk + 8 * lh;
    v8h a0 = *(const v8h*)(Arow + base);
    v8h a1 = *(const v8h*)(Arow + base + 16);
    v8h b0 = *(const v8h*)(Brow + base);
    v8h b1 = *(const v8h*)(Brow + base + 16);
    acc = wmma_step(a0, a1, b0, b1, acc);
  }
  const float s = 0.044194173824159216f;  // 1/sqrt(512)
#pragma unroll
  for (int r2 = 0; r2 < 8; ++r2) {
    int c = lh * 8 + r2;
    S[b * 256 + c * 16 + lm] = acc[r2] * s;
  }
}

// ---------------------------------------------------------------------------
// Legacy nn.Softmax() on 3D: softmax over dim 0 (batch). 256 independent
// length-128 softmaxes, one thread each. In-place on S.
// ---------------------------------------------------------------------------
__global__ void k_softmax_batch(float* __restrict__ S) {
  int j = threadIdx.x;  // (c*16 + d)
  float mx = -1e30f;
  for (int b = 0; b < BATCH; ++b) mx = fmaxf(mx, S[b * 256 + j]);
  float sum = 0.0f;
  for (int b = 0; b < BATCH; ++b) sum += __expf(S[b * 256 + j] - mx);
  float inv = 1.0f / sum;
  for (int b = 0; b < BATCH; ++b)
    S[b * 256 + j] = __expf(S[b * 256 + j] - mx) * inv;
}

// ---------------------------------------------------------------------------
// e = relu(LayerNorm_{16,512}( ab[b] @ a[b] )) per batch; one block per b.
// ---------------------------------------------------------------------------
__global__ void __launch_bounds__(256)
k_context_ln(const float* __restrict__ AB, const float* __restrict__ Af,
             const float* __restrict__ g, const float* __restrict__ beta,
             half_t* __restrict__ Eh) {
  __shared__ float ab[256];
  __shared__ float ep[8192];
  __shared__ float rs[256], rq[256];
  const int b = blockIdx.x, tid = threadIdx.x;
  ab[tid] = AB[b * 256 + tid];
  __syncthreads();

  float lsum = 0.0f, lsq = 0.0f;
#pragma unroll 4
  for (int i = 0; i < 32; ++i) {
    int idx = tid + 256 * i;          // idx = c*512 + t
    int c = idx >> 9, t = idx & 511;
    float s = 0.0f;
#pragma unroll
    for (int d = 0; d < 16; ++d)
      s += ab[c * 16 + d] * Af[(size_t)(b * 16 + d) * D1 + t];
    ep[idx] = s;
    lsum += s;
    lsq += s * s;
  }
  rs[tid] = lsum; rq[tid] = lsq;
  __syncthreads();
  for (int s2 = 128; s2 > 0; s2 >>= 1) {
    if (tid < s2) { rs[tid] += rs[tid + s2]; rq[tid] += rq[tid + s2]; }
    __syncthreads();
  }
  float mu   = rs[0] * (1.0f / 8192.0f);
  float var  = rq[0] * (1.0f / 8192.0f) - mu * mu;
  float rstd = rsqrtf(var + 1e-5f);
#pragma unroll 4
  for (int i = 0; i < 32; ++i) {
    int idx = tid + 256 * i;
    int c = idx >> 9, t = idx & 511;
    float v = (ep[idx] - mu) * rstd * g[idx] + beta[idx];
    v = fmaxf(v, 0.0f);
    Eh[(size_t)(b * 16 + c) * D1 + t] = (half_t)v;
  }
}

// ---------------------------------------------------------------------------
// h1 = relu( feat[128,16384] @ w1[200,16384]^T + b1 ).  feat = [xp | new_x]
// ---------------------------------------------------------------------------
__global__ void __launch_bounds__(32)
k_gemm_cls(const half_t* __restrict__ XPh, const half_t* __restrict__ NXh,
           const half_t* __restrict__ W1h, const float* __restrict__ b1,
           float* __restrict__ H1) {
  const int lane = threadIdx.x & 31;
  const int lm = lane & 15, lh = lane >> 4;
  const int m0 = blockIdx.x * 16;    // batch tile (8 tiles)
  const int ntile = blockIdx.y;      // 13 tiles cover N=200 (pad to 208)
  int n = ntile * 16 + lm;
  int nl = (n < N1) ? n : (N1 - 1);  // clamp loads; masked at store
  const half_t* Wrow = W1h + (size_t)nl * KCLS;
  const int mrow = m0 + lm;
  const half_t* X0 = XPh + (size_t)mrow * 8192;
  const half_t* X1 = NXh + (size_t)mrow * 8192;

  v8f acc = {};
  for (int kk = 0; kk < KCLS; kk += 32) {
    if (kk + 128 < KCLS) __builtin_prefetch(Wrow + kk + 128, 0, 3);
    const half_t* Xs = (kk < 8192) ? (X0 + kk) : (X1 + (kk - 8192));
    int base8 = 8 * lh;
    v8h a0 = *(const v8h*)(Xs + base8);
    v8h a1 = *(const v8h*)(Xs + base8 + 16);
    v8h b0 = *(const v8h*)(Wrow + kk + base8);
    v8h b1 = *(const v8h*)(Wrow + kk + base8 + 16);
    acc = wmma_step(a0, a1, b0, b1, acc);
  }
  if (n < N1) {
    float bn = b1[n];
#pragma unroll
    for (int r2 = 0; r2 < 8; ++r2) {
      int m = m0 + lh * 8 + r2;
      H1[m * N1 + n] = fmaxf(acc[r2] + bn, 0.0f);
    }
  }
}

// ---------------------------------------------------------------------------
// Tiny tail FCs (200->50->10->128): one thread per output, f32 VALU.
// ---------------------------------------------------------------------------
__global__ void k_fc(const float* __restrict__ X, const float* __restrict__ W,
                     const float* __restrict__ bias, float* __restrict__ Y,
                     int M, int N, int K, int relu) {
  int idx = blockIdx.x * blockDim.x + threadIdx.x;
  if (idx >= M * N) return;
  int m = idx / N, n = idx % N;
  const float* xr = X + (size_t)m * K;
  const float* wr = W + (size_t)n * K;
  float s = bias[n];
  for (int k = 0; k < K; ++k) s += xr[k] * wr[k];
  if (relu) s = fmaxf(s, 0.0f);
  Y[idx] = s;
}

// ---------------------------------------------------------------------------
extern "C" void kernel_launch(void* const* d_in, const int* in_sizes, int n_in,
                              void* d_out, int out_size, void* d_ws, size_t ws_size,
                              hipStream_t stream) {
  (void)in_sizes; (void)n_in; (void)out_size; (void)ws_size;
  const float* x      = (const float*)d_in[0];
  const float* attn_w = (const float*)d_in[1];
  const float* attn_b = (const float*)d_in[2];
  const float* ln_g   = (const float*)d_in[3];
  const float* ln_b   = (const float*)d_in[4];
  const float* w1     = (const float*)d_in[5];
  const float* b1     = (const float*)d_in[6];
  const float* w2     = (const float*)d_in[7];
  const float* b2     = (const float*)d_in[8];
  const float* w3     = (const float*)d_in[9];
  const float* b3     = (const float*)d_in[10];
  const float* w4     = (const float*)d_in[11];
  const float* b4     = (const float*)d_in[12];
  float* out = (float*)d_out;

  char* ws = (char*)d_ws;
  size_t off = 0;
  auto alloc = [&](size_t bytes) -> char* {
    char* p = ws + off;
    off += (bytes + 255) & ~(size_t)255;
    return p;
  };
  float*  xp_f  = (float*) alloc((size_t)M_ROWS * D1 * 4);
  half_t* xp_h  = (half_t*)alloc((size_t)M_ROWS * D1 * 2);
  float*  nx_f  = (float*) alloc((size_t)M_ROWS * D1 * 4);
  half_t* nx_h  = (half_t*)alloc((size_t)M_ROWS * D1 * 2);
  float*  a_f   = (float*) alloc((size_t)M_ROWS * D1 * 4);
  half_t* a_h   = (half_t*)alloc((size_t)M_ROWS * D1 * 2);
  half_t* bp_h  = (half_t*)alloc((size_t)M_ROWS * D1 * 2);
  half_t* e_h   = (half_t*)alloc((size_t)M_ROWS * D1 * 2);
  float*  sc    = (float*) alloc((size_t)BATCH * 256 * 4);
  half_t* awh   = (half_t*)alloc((size_t)2 * 4 * D1 * D1 * 2);
  half_t* w1h   = (half_t*)alloc((size_t)N1 * KCLS * 2);
  float*  h1    = (float*) alloc((size_t)BATCH * N1 * 4);
  float*  h2    = (float*) alloc((size_t)BATCH * 50 * 4);
  float*  h3    = (float*) alloc((size_t)BATCH * 10 * 4);

  // Weight conversions (L2-resident afterwards).
  {
    int n = 2 * 4 * D1 * D1;  // 2,097,152
    k_f32_to_f16<<<(n + 255) / 256, 256, 0, stream>>>(attn_w, awh, n);
    n = N1 * KCLS;            // 3,276,800
    k_f32_to_f16<<<(n + 255) / 256, 256, 0, stream>>>(w1, w1h, n);
  }

  // Max-pool (the HBM-bound stage) + new_x init.
  k_pool<<<(M_ROWS * D1 + 255) / 256, 256, 0, stream>>>(x, xp_f, xp_h, nx_f, nx_h);

  dim3 ggrid(M_ROWS / 16, 2);  // 128 M-tiles x 2 groups of 8 waves x 32 cols
  for (int i = 0; i < 2; ++i) {
    const half_t* Wa  = awh + (size_t)(i * 4 + 0) * D1 * D1;
    const half_t* Wb  = awh + (size_t)(i * 4 + 1) * D1 * D1;
    const half_t* W1s = awh + (size_t)(i * 4 + 3) * D1 * D1;
    const float* ba  = attn_b + (size_t)(i * 4 + 0) * D1;
    const float* bb  = attn_b + (size_t)(i * 4 + 1) * D1;
    const float* b1s = attn_b + (size_t)(i * 4 + 3) * D1;

    // a = fc_a(new_x)  (f32 for context matmul, f16 for scores)
    k_gemm512<<<ggrid, 256, 0, stream>>>(nx_h, Wa, ba, nullptr, a_f, a_h, 0);
    // bproj = fc_b(xp)
    k_gemm512<<<ggrid, 256, 0, stream>>>(xp_h, Wb, bb, nullptr, nullptr, bp_h, 0);
    // scores = a @ bproj^T / sqrt(512)
    k_scores<<<BATCH / 4, 128, 0, stream>>>(a_h, bp_h, sc);
    // softmax over the batch axis (legacy nn.Softmax on 3D)
    k_softmax_batch<<<1, 256, 0, stream>>>(sc);
    // e = relu(LN(ab @ a))
    k_context_ln<<<BATCH, 256, 0, stream>>>(sc, a_f, ln_g, ln_b, e_h);
    // new_x = new_x + fc_1(e)
    k_gemm512<<<ggrid, 256, 0, stream>>>(e_h, W1s, b1s, nx_f, nx_f, nx_h, 0);
  }

  // Classifier head.
  k_gemm_cls<<<dim3(BATCH / 16, 13), 32, 0, stream>>>(xp_h, nx_h, w1h, b1, h1);
  k_fc<<<(BATCH * 50 + 255) / 256, 256, 0, stream>>>(h1, w2, b2, h2, BATCH, 50, N1, 1);
  k_fc<<<(BATCH * 10 + 255) / 256, 256, 0, stream>>>(h2, w3, b3, h3, BATCH, 10, 50, 1);
  k_fc<<<(BATCH * 128 + 255) / 256, 256, 0, stream>>>(h3, w4, b4, out, BATCH, 128, 10, 0);
}
